// TSCN_80032420593738
// MI455X (gfx1250) — matrix operand order repaired and checked by reading, not compiled, not run on hardware
//
#include <hip/hip_runtime.h>

typedef __attribute__((ext_vector_type(16))) _Float16 v16h;
typedef __attribute__((ext_vector_type(8)))  _Float16 h8;
typedef __attribute__((ext_vector_type(8)))  float    v8f;
typedef __attribute__((ext_vector_type(2)))  _Float16 h2;
typedef __attribute__((ext_vector_type(2)))  float    f2;

#define DIM  64
#define SS   32
#define HIST 50
#define XPAD 136   // padded row stride (halfs): 272B = 68 banks -> conflict-free row access

// ---------------------------------------------------------------- max(n_idxs)
__global__ __launch_bounds__(1024)
void max_kernel(const int* __restrict__ n_idxs, int B, float* __restrict__ maxp) {
    __shared__ int red[1024];
    int tid = threadIdx.x;
    int m = 1;
    for (int i = tid; i < B; i += 1024) m = max(m, n_idxs[i]);
    red[tid] = m; __syncthreads();
    for (int o = 512; o > 0; o >>= 1) {
        if (tid < o) red[tid] = max(red[tid], red[tid + o]);
        __syncthreads();
    }
    if (tid == 0) maxp[0] = (float)red[0];
}

// ---------------------------------------------------------------- user embedding
__global__ __launch_bounds__(64)
void user_kernel(const int* __restrict__ user_inputs, const int* __restrict__ n_idxs,
                 const float* __restrict__ emb, const float* __restrict__ maxp,
                 float* __restrict__ user_emb) {
    int b = blockIdx.x;
    int d = threadIdx.x;
    int n = n_idxs[b];
    float acc = 0.f;
    for (int t = 0; t < n; ++t) {
        int idx = user_inputs[b * HIST + t];
        acc += emb[(size_t)idx * DIM + d];
    }
    user_emb[(size_t)b * DIM + d] = acc / maxp[0];
}

// ---------------------------------------------------------------- adamic pool (all 3 uses)
// mode 0: parent=item[row];               self=emb[parent]; neigh/a from adj tables
// mode 1: parent=adj_item[item[row/32],row%32]; self=emb[parent]; neigh/a from adj tables
// mode 2: self=self_direct[row]; neigh=neigh_direct[row*32+s]; a=adj_adam[item[row],s]
__global__ __launch_bounds__(512)
void pool_kernel(int mode,
                 const int*   __restrict__ item_inputs,
                 const int*   __restrict__ adj_item,
                 const float* __restrict__ adj_adam,
                 const float* __restrict__ emb,
                 const float* __restrict__ self_direct,
                 const float* __restrict__ neigh_direct,
                 const float* __restrict__ pool_w,
                 const float* __restrict__ pool_b,
                 float* __restrict__ out,
                 int nRows)
{
    // pool_w staged TRANSPOSED: WhT[n][k], so a WMMA B-fragment (one column,
    // two runs of 8 consecutive K) is two contiguous ds_load_b128.
    __shared__ _Float16 WhT[64 * XPAD];
    __shared__ _Float16 Xs [16 * XPAD];   // 16 activation rows, 128 cols used

    const int tid  = threadIdx.x;
    const int wave = tid >> 5;
    const int lane = tid & 31;

    // stage weights (transposed) once per block
    for (int i = tid; i < 128 * 64; i += 512) {
        int k = i >> 6, n = i & 63;
        WhT[n * XPAD + k] = (_Float16)pool_w[i];
    }

    const int row = blockIdx.x * 16 + wave;   // one wave builds one activation row

    if (row < nRows) {
        const float* selfp;
        int parent;
        if (mode == 0) {
            parent = item_inputs[row];
            selfp  = emb + (size_t)parent * DIM;
        } else if (mode == 1) {
            int b = row >> 5, n = row & 31;
            parent = adj_item[(size_t)item_inputs[b] * SS + n];
            selfp  = emb + (size_t)parent * DIM;
        } else {
            parent = item_inputs[row];
            selfp  = self_direct + (size_t)row * DIM;
        }

        // per-lane adamic weight (+ neighbor index for gather modes)
        float a = adj_adam[(size_t)parent * SS + lane];
        int nidx = (mode == 2) ? 0 : adj_item[(size_t)parent * SS + lane];

        // softmax over the 32 lanes (wave32 butterfly)
        float m = a;
        for (int o = 16; o > 0; o >>= 1) m = fmaxf(m, __shfl_xor(m, o, 32));
        float e = __expf(a - m);
        float ssum = e;
        for (int o = 16; o > 0; o >>= 1) ssum += __shfl_xor(ssum, o, 32);
        float w = e / ssum;

        // self row: lane holds cols 2*lane, 2*lane+1 (coalesced 256B)
        f2 self2 = ((const f2*)selfp)[lane];

        // weighted aggregation: broadcast neighbor s, whole wave loads its 256B row
        f2 acc; acc.x = 0.f; acc.y = 0.f;
        for (int s = 0; s < SS; ++s) {
            float ws_ = __shfl(w, s, 32);
            const float* np;
            if (mode == 2) {
                np = neigh_direct + ((size_t)row * SS + s) * DIM;
            } else {
                int ni = __shfl(nidx, s, 32);
                np = emb + (size_t)ni * DIM;
            }
            f2 nv = ((const f2*)np)[lane];
            acc.x += ws_ * nv.x;
            acc.y += ws_ * nv.y;
        }

        // x = [self(0..63) | agg(64..127)] -> f16 into LDS
        h2* xrow = (h2*)&Xs[wave * XPAD];
        h2 sh; sh.x = (_Float16)self2.x; sh.y = (_Float16)self2.y;
        h2 ah; ah.x = (_Float16)acc.x;   ah.y = (_Float16)acc.y;
        xrow[lane]      = sh;
        xrow[32 + lane] = ah;
    }
    __syncthreads();

    // phase 2: 4 waves compute Y[16x64] = X[16x128] * W[128x64] via v_wmma_f32_16x16x32_f16
    if (wave < 4) {
        const int nt   = wave;        // 16-col output tile
        const int half = lane >> 4;   // lane half: A K+8 / B K+8 / C M+8
        const int lp   = lane & 15;
        const int col  = nt * 16 + lp;

        float bias = pool_b[col];     // hoisted: issued before WMMAs, overlaps latency

        v8f c = {};
        for (int kc = 0; kc < 4; ++kc) {   // K = 128 in chunks of 32
            // A 16x32 f16 layout: lane lp = row M; elems 0-7 = K kc*32+8*half+{0..7},
            //                                      elems 8-15 = K kc*32+16+8*half+{0..7}
            h8 a0 = *(const h8*)&Xs[lp * XPAD + kc * 32 +      8 * half];
            h8 a1 = *(const h8*)&Xs[lp * XPAD + kc * 32 + 16 + 8 * half];
            // B 32x16 f16 layout: lane lp = col N; elems 0-7 = K kc*32+8*half+{0..7},
            //                                      elems 8-15 = K kc*32+16+8*half+{0..7}
            h8 b0 = *(const h8*)&WhT[col * XPAD + kc * 32 +      8 * half];
            h8 b1 = *(const h8*)&WhT[col * XPAD + kc * 32 + 16 + 8 * half];
            v16h af, bf;
#pragma unroll
            for (int j = 0; j < 8; ++j) {
                af[j] = a0[j]; af[8 + j] = a1[j];
                bf[j] = b0[j]; bf[8 + j] = b1[j];
            }
            c = __builtin_amdgcn_wmma_f32_16x16x32_f16(
                    false, af, false, bf, (short)0, c, false, false);
        }

        // C/D layout: VGPR r holds M = r + 8*half, N = lane&15.
        // Base pointer hoisted; per-row store is a constant-offset global_store_b32.
        const int baseRow = blockIdx.x * 16 + 8 * half;
        float* op = out + (size_t)baseRow * DIM + col;
        const int rowsLeft = nRows - baseRow;   // uniform per 16-lane half
#pragma unroll
        for (int r = 0; r < 8; ++r) {
            if (r < rowsLeft) {
                float v = c[r] + bias;
                op[(size_t)r * DIM] = v > 0.f ? v : 0.f;
            }
        }
    }
}

// ---------------------------------------------------------------- final MLP -> logits
__global__ __launch_bounds__(256)
void fc_kernel(const float* __restrict__ user_emb, const float* __restrict__ item_emb,
               const float* __restrict__ fc1_w, const float* __restrict__ fc1_b,
               const float* __restrict__ fc2_w, const float* __restrict__ fc2_b,
               float* __restrict__ logits, int B)
{
    __shared__ float xsh[8 * 128];
    int wave = threadIdx.x >> 5, lane = threadIdx.x & 31;
    int b = blockIdx.x * 8 + wave;
    bool ok = b < B;
    for (int i = lane; i < 64; i += 32) {
        xsh[wave * 128 + i]      = ok ? user_emb[(size_t)b * 64 + i] : 0.f;
        xsh[wave * 128 + 64 + i] = ok ? item_emb[(size_t)b * 64 + i] : 0.f;
    }
    __syncthreads();
    float h = fc1_b[lane];
    for (int k = 0; k < 128; ++k)
        h += xsh[wave * 128 + k] * fc1_w[k * 32 + lane];
    h = h > 0.f ? h : 0.f;
    float o = h * fc2_w[lane];
    for (int off = 16; off > 0; off >>= 1) o += __shfl_xor(o, off, 32);
    if (ok && lane == 0) logits[b] = o + fc2_b[0];
}

// ---------------------------------------------------------------- softmax over B=2048 logits
__global__ __launch_bounds__(1024)
void softmax_kernel(const float* __restrict__ logits, float* __restrict__ out, int B) {
    __shared__ float red[1024];
    int tid = threadIdx.x;
    float l0 = (tid < B) ? logits[tid] : -1e30f;
    float l1 = (tid + 1024 < B) ? logits[tid + 1024] : -1e30f;
    red[tid] = fmaxf(l0, l1); __syncthreads();
    for (int o = 512; o > 0; o >>= 1) {
        if (tid < o) red[tid] = fmaxf(red[tid], red[tid + o]);
        __syncthreads();
    }
    float gm = red[0]; __syncthreads();
    float e0 = __expf(l0 - gm), e1 = __expf(l1 - gm);
    red[tid] = ((tid < B) ? e0 : 0.f) + ((tid + 1024 < B) ? e1 : 0.f);
    __syncthreads();
    for (int o = 512; o > 0; o >>= 1) {
        if (tid < o) red[tid] += red[tid + o];
        __syncthreads();
    }
    float inv = 1.f / red[0];
    if (tid < B)        out[tid]        = e0 * inv;
    if (tid + 1024 < B) out[tid + 1024] = e1 * inv;
}

// ----------------------------------------------------------------
extern "C" void kernel_launch(void* const* d_in, const int* in_sizes, int n_in,
                              void* d_out, int out_size, void* d_ws, size_t ws_size,
                              hipStream_t stream) {
    const int*   user_inputs = (const int*)  d_in[0];
    const int*   item_inputs = (const int*)  d_in[1];
    const int*   n_idxs      = (const int*)  d_in[2];
    const float* emb         = (const float*)d_in[3];
    const int*   adj_item    = (const int*)  d_in[4];
    const float* adj_adam    = (const float*)d_in[5];
    const float* pool_w      = (const float*)d_in[6];
    const float* pool_b      = (const float*)d_in[7];
    const float* fc1_w       = (const float*)d_in[8];
    const float* fc1_b       = (const float*)d_in[9];
    const float* fc2_w       = (const float*)d_in[10];
    const float* fc2_b       = (const float*)d_in[11];
    const int B = in_sizes[1];           // 2048

    float* ws       = (float*)d_ws;
    float* maxp     = ws;                              // [64] (1 used)
    float* user_emb = ws + 64;                         // [B*64]
    float* v0       = user_emb + (size_t)B * 64;       // [B*64]
    float* item_emb = v0       + (size_t)B * 64;       // [B*64]
    float* logits   = item_emb + (size_t)B * 64;       // [B]
    float* v1       = logits   + B;                    // [B*32*64]

    max_kernel <<<1, 1024, 0, stream>>>(n_idxs, B, maxp);
    user_kernel<<<B, 64,  0, stream>>>(user_inputs, n_idxs, emb, maxp, user_emb);

    // hop-1 pool: 65536 rows -> v1
    pool_kernel<<<(B * SS + 15) / 16, 512, 0, stream>>>(
        1, item_inputs, adj_item, adj_adam, emb, nullptr, nullptr,
        pool_w, pool_b, v1, B * SS);
    // hop-0 pool (first pass): B rows -> v0
    pool_kernel<<<(B + 15) / 16, 512, 0, stream>>>(
        0, item_inputs, adj_item, adj_adam, emb, nullptr, nullptr,
        pool_w, pool_b, v0, B);
    // hop-0 pool (second pass, inputs v0/v1): B rows -> item_emb
    pool_kernel<<<(B + 15) / 16, 512, 0, stream>>>(
        2, item_inputs, adj_item, adj_adam, emb, v0, v1,
        pool_w, pool_b, item_emb, B);

    fc_kernel<<<(B + 7) / 8, 256, 0, stream>>>(
        user_emb, item_emb, fc1_w, fc1_b, fc2_w, fc2_b, logits, B);
    softmax_kernel<<<1, 1024, 0, stream>>>(logits, (float*)d_out, B);
}
